// BaselineModel_33775622815821
// MI455X (gfx1250) — matrix operand order repaired and checked by reading, not compiled, not run on hardware
//
#include <hip/hip_runtime.h>
#include <hip/hip_bf16.h>
#include <math.h>

typedef __bf16 bf16;
typedef __attribute__((ext_vector_type(16))) __bf16 bf16x16;
typedef __attribute__((ext_vector_type(8)))  __bf16 bf16x8;
typedef __attribute__((ext_vector_type(8)))  float  f32x8;

#define B_   2
#define S_   2048
#define H_   1024
#define NH_  16
#define HD_  64
#define L_   4
#define ROWS_ (B_*S_)   // 4096

union AFrag { bf16x16 v; bf16x8 h[2]; bf16 e[16]; };
union CFrag { f32x8 v; float e[8]; };

__device__ __forceinline__ float sigmf(float x) { return 1.0f / (1.0f + __expf(-x)); }

// ---------------------------------------------------------------------------
// f32 -> bf16 conversion into B-fragment-major ("swizzled") layout.
// Fragment order for a KxN matrix used as the WMMA B operand:
//   flat = ((kc*(N/16) + nc)*32 + lane)*16 + e
//   where kc=K/32 chunk, nc=N/16 chunk, lane = (Klocal>=16 ? 16:0) + (n%16),
//         e = Klocal%16.
// Each lane's 16 fragment elements are contiguous (32B); a wave's fragment is
// 1KB contiguous -> two global_load_b128 per fragment, fully coalesced.
// ---------------------------------------------------------------------------
__global__ void f2b_swz(const float* __restrict__ in, bf16* __restrict__ outp,
                        int K, int N) {
    int i = blockIdx.x * blockDim.x + threadIdx.x;   // dest-indexed
    if (i >= K * N) return;
    int e    = i & 15;
    int lane = (i >> 4) & 31;
    int nch  = N >> 4;
    int nc   = (i >> 9) % nch;
    int kc   = (i >> 9) / nch;
    int k = kc * 32 + (lane & 16) + e;
    int n = nc * 16 + (lane & 15);
    outp[i] = (bf16)in[(size_t)k * N + n];
}

// ---------------------------------------------------------------------------
// Embedding gather (optionally masked: id = (msk==sel) ? ids : 0) -> bf16 rows
// ---------------------------------------------------------------------------
__global__ void gather_rows(const int* __restrict__ ids, const int* __restrict__ msk,
                            int sel, const float* __restrict__ table,
                            bf16* __restrict__ outp, int nrows) {
    int i = blockIdx.x * blockDim.x + threadIdx.x;
    if (i >= nrows * H_) return;
    int row = i >> 10;
    int c   = i & (H_ - 1);
    int id = ids[row];
    if (msk) id = (msk[row] == sel) ? id : 0;
    outp[i] = (bf16)table[(size_t)id * H_ + c];
}

// ---------------------------------------------------------------------------
// WMMA GEMM: C[M,N] = act( A(bf16 row-major, MxK) @ Wswz(bf16 fragment-major)
//                          + bias + residual )
// 256 threads = 8 waves; block tile 32(M) x 256(N); wave tile 16 x 64.
// No LDS, no barriers: per k-step = 2 A b128 loads + 8 B b128 loads + 4 WMMA.
// ---------------------------------------------------------------------------
__global__ __launch_bounds__(256)
void gemm_bf16(const bf16* __restrict__ A, const bf16* __restrict__ W,
               const float* __restrict__ bias, const float* __restrict__ residual,
               float* __restrict__ C, bf16* __restrict__ Cb,
               int M, int N, int K, int act) {
    const int tid  = threadIdx.x;
    const int lane = tid & 31;
    const int wid  = tid >> 5;
    const int n_blk = blockIdx.x * 256;
    const int m0    = blockIdx.y * 32 + (wid & 1) * 16;
    const int nw    = (wid >> 1) * 64;               // 0,64,128,192 in block tile
    const int nchunks = N >> 4;

    CFrag acc[4];
    #pragma unroll
    for (int j = 0; j < 4; ++j)
        #pragma unroll
        for (int r = 0; r < 8; ++r) acc[j].e[r] = 0.0f;

    const int arow = m0 + (lane & 15);
    const int kb   = (lane < 16) ? 0 : 16;           // A-fragment K base per lane half
    const bf16* aptr = A + (size_t)arow * K + kb;
    const bf16* wptr = W + ((size_t)((n_blk + nw) >> 4)) * 512 + lane * 16;
    const size_t wstep = (size_t)nchunks * 512;

    for (int k0 = 0; k0 < K; k0 += 32) {
        AFrag a;
        a.h[0] = *(const bf16x8*)(aptr);
        a.h[1] = *(const bf16x8*)(aptr + 8);
        aptr += 32;
        __builtin_prefetch(wptr + wstep, 0, 1);
        #pragma unroll
        for (int j = 0; j < 4; ++j) {
            AFrag b;
            const bf16* bp = wptr + j * 512;
            b.h[0] = *(const bf16x8*)(bp);
            b.h[1] = *(const bf16x8*)(bp + 8);
            acc[j].v = __builtin_amdgcn_wmma_f32_16x16x32_bf16(
                false, a.v, false, b.v, (short)0, acc[j].v, false, false);
        }
        wptr += wstep;
    }

    const int mro  = (lane < 16) ? 0 : 8;
    const int colb = lane & 15;
    #pragma unroll
    for (int j = 0; j < 4; ++j) {
        #pragma unroll
        for (int r = 0; r < 8; ++r) {
            int m = m0 + r + mro;
            int n = n_blk + nw + j * 16 + colb;
            size_t o = (size_t)m * N + n;
            float v = acc[j].e[r] + bias[n];
            if (residual) v += residual[o];
            if (act) v = fmaxf(v, 0.0f);
            C[o] = v;
            if (Cb) Cb[o] = (bf16)v;
        }
    }
}

// ---------------------------------------------------------------------------
// x = (all_item + all_user) * sqrt(H);  also bf16 mirror
// ---------------------------------------------------------------------------
__global__ void combine_kernel(const float* __restrict__ ai, const float* __restrict__ au,
                               float* __restrict__ x, bf16* __restrict__ xb) {
    int i = blockIdx.x * blockDim.x + threadIdx.x;
    if (i >= ROWS_ * H_) return;
    float v = (ai[i] + au[i]) * 32.0f;   // sqrt(1024)
    x[i] = v;
    xb[i] = (bf16)v;
}

// ---------------------------------------------------------------------------
// RoPE on Q,K + head-major repack -> bf16.
// Qr, Kr: [B,NH,S,HD] row-major (d contiguous) — score B-fragments are
//         contiguous 16-element runs there.
// Vh: fragment-major per head for the o-accumulation B operand:
//         flat = ((bh*(S/32) + kc)*4 + nc)*512 + lane*16 + e
//         lane = (key%32>=16 ? 16:0) + (d%16),  e = key%16,  nc = d/16.
// ---------------------------------------------------------------------------
__global__ void rope_split(const float* __restrict__ proj,
                           bf16* __restrict__ Qr, bf16* __restrict__ Kr,
                           bf16* __restrict__ Vh) {
    int i = blockIdx.x * blockDim.x + threadIdx.x;  // over B*NH*S*32
    if (i >= B_ * NH_ * S_ * 32) return;
    int d = i & 31;
    int s = (i >> 5) & (S_ - 1);
    int h = (i >> 16) & (NH_ - 1);
    int b = i >> 20;
    size_t prow = ((size_t)(b * S_ + s)) * (4 * H_);
    int c = h * HD_ + d;
    float q1 = proj[prow + 2 * H_ + c], q2 = proj[prow + 2 * H_ + c + 32];
    float k1 = proj[prow + 3 * H_ + c], k2 = proj[prow + 3 * H_ + c + 32];
    float v1 = proj[prow + 1 * H_ + c], v2 = proj[prow + 1 * H_ + c + 32];
    float inv = __powf(10000.0f, -(float)d * (1.0f / 32.0f));
    float sn, cs;
    __sincosf((float)s * inv, &sn, &cs);
    int bh = b * NH_ + h;
    size_t o = (((size_t)bh) * S_ + s) * HD_ + d;
    Qr[o]      = (bf16)(q1 * cs - q2 * sn);
    Qr[o + 32] = (bf16)(q2 * cs + q1 * sn);
    Kr[o]      = (bf16)(k1 * cs - k2 * sn);
    Kr[o + 32] = (bf16)(k2 * cs + k1 * sn);
    // swizzled V
    size_t vbase = (((size_t)bh * (S_ / 32) + (s >> 5)) * 4) * 512
                 + (size_t)((s & 16) + (d & 15)) * 16 + (s & 15);
    Vh[vbase + (size_t)(d >> 4) * 512]        = (bf16)v1;   // col d
    Vh[vbase + (size_t)((d + 32) >> 4) * 512] = (bf16)v2;   // col d+32
}

// ---------------------------------------------------------------------------
// HSTU sigmoid attention. 128 threads = 4 waves; one wave per 16-row Q tile
// of one head. Per 32-key chunk: 4 WMMAs (scores, K-dim = head dim) ->
// sigmoid+causal+pad mask in C fragment -> per-wave LDS re-stripe to A layout
// -> 4 WMMAs (o += w.V, V pre-swizzled). Output written in [B,S,H].
// ---------------------------------------------------------------------------
__global__ __launch_bounds__(128)
void hstu_attn(const bf16* __restrict__ Q, const bf16* __restrict__ Kr,
               const bf16* __restrict__ V, const int* __restrict__ smask,
               float* __restrict__ O) {
    __shared__ bf16 sS[4 * 16 * 32];
    const int tid  = threadIdx.x;
    const int lane = tid & 31;
    const int wid  = tid >> 5;
    const int t  = blockIdx.x * 4 + wid;     // 0..4095
    const int qt = t & 127;
    const int bh = t >> 7;                   // 0..31
    const int b  = bh >> 4;
    const int h  = bh & 15;
    const int q0 = qt * 16;

    const bf16* Qbase = Q  + (size_t)bh * S_ * HD_;
    const bf16* Kbase = Kr + (size_t)bh * S_ * HD_;
    const bf16* Vbase = V  + (size_t)bh * (S_ / 32) * 4 * 512;

    const int rlow = lane & 15;
    const int kb   = (lane < 16) ? 0 : 16;
    const int kb2  = (lane < 16) ? 0 : 16;
    const int mro  = (lane < 16) ? 0 : 8;

    AFrag aq0, aq1;
    {
        const bf16* qrow = Qbase + (size_t)(q0 + rlow) * HD_;
        aq0.h[0] = *(const bf16x8*)(qrow + kb);
        aq0.h[1] = *(const bf16x8*)(qrow + kb + 8);
        aq1.h[0] = *(const bf16x8*)(qrow + 32 + kb);
        aq1.h[1] = *(const bf16x8*)(qrow + 32 + kb + 8);
    }

    CFrag oacc[4];
    #pragma unroll
    for (int j = 0; j < 4; ++j)
        #pragma unroll
        for (int r = 0; r < 8; ++r) oacc[j].e[r] = 0.0f;

    bf16* myS = sS + wid * (16 * 32);
    const int qmax = q0 + 15;

    for (int k0 = 0; k0 <= qmax; k0 += 32) {
        #pragma unroll
        for (int sub = 0; sub < 2; ++sub) {
            const int key0 = k0 + sub * 16;
            const bf16* krow = Kbase + (size_t)(key0 + rlow) * HD_;  // lane's key column
            AFrag b0, b1;
            b0.h[0] = *(const bf16x8*)(krow + kb2);
            b0.h[1] = *(const bf16x8*)(krow + kb2 + 8);
            b1.h[0] = *(const bf16x8*)(krow + 32 + kb2);
            b1.h[1] = *(const bf16x8*)(krow + 32 + kb2 + 8);

            CFrag sc;
            #pragma unroll
            for (int r = 0; r < 8; ++r) sc.e[r] = 0.0f;
            sc.v = __builtin_amdgcn_wmma_f32_16x16x32_bf16(
                false, aq0.v, false, b0.v, (short)0, sc.v, false, false);
            sc.v = __builtin_amdgcn_wmma_f32_16x16x32_bf16(
                false, aq1.v, false, b1.v, (short)0, sc.v, false, false);

            const int key = key0 + rlow;
            const int mk  = smask[b * S_ + key];
            #pragma unroll
            for (int r = 0; r < 8; ++r) {
                int q = q0 + r + mro;
                float w = (key <= q && mk != 0) ? sigmf(sc.e[r] * 0.125f) : 0.0f;
                myS[(r + mro) * 32 + sub * 16 + rlow] = (bf16)w;
            }
        }
        asm volatile("" ::: "memory");  // keep w stores before re-striped loads

        AFrag aw;
        {
            const bf16* wrow = myS + rlow * 32;
            aw.h[0] = *(const bf16x8*)(wrow + kb);
            aw.h[1] = *(const bf16x8*)(wrow + kb + 8);
        }

        const bf16* vk = Vbase + ((size_t)(k0 >> 5) * 4) * 512 + lane * 16;
        #pragma unroll
        for (int j = 0; j < 4; ++j) {
            AFrag bv;
            const bf16* bp = vk + j * 512;
            bv.h[0] = *(const bf16x8*)(bp);
            bv.h[1] = *(const bf16x8*)(bp + 8);
            oacc[j].v = __builtin_amdgcn_wmma_f32_16x16x32_bf16(
                false, aw.v, false, bv.v, (short)0, oacc[j].v, false, false);
        }
    }

    #pragma unroll
    for (int j = 0; j < 4; ++j)
        #pragma unroll
        for (int r = 0; r < 8; ++r) {
            int q = q0 + r + mro;
            O[((size_t)(b * S_ + q)) * H_ + h * HD_ + j * 16 + rlow] = oacc[j].e[r];
        }
}

// ---------------------------------------------------------------------------
// o_ln = LN(o)*g + b;  gated = silu(U) * o_ln  (bf16).  One block per row.
// ---------------------------------------------------------------------------
__global__ __launch_bounds__(256)
void ln_gate(const float* __restrict__ O, const float* __restrict__ proj,
             const float* __restrict__ g, const float* __restrict__ bta,
             bf16* __restrict__ gated) {
    __shared__ float red[256];
    const int row = blockIdx.x;
    const int tid = threadIdx.x;
    const float* o = O + (size_t)row * H_;
    float vals[4], s = 0.0f;
    #pragma unroll
    for (int u = 0; u < 4; ++u) { vals[u] = o[tid + u * 256]; s += vals[u]; }
    red[tid] = s; __syncthreads();
    for (int st = 128; st > 0; st >>= 1) { if (tid < st) red[tid] += red[tid + st]; __syncthreads(); }
    const float mean = red[0] * (1.0f / H_);
    __syncthreads();
    s = 0.0f;
    #pragma unroll
    for (int u = 0; u < 4; ++u) { float d = vals[u] - mean; s += d * d; }
    red[tid] = s; __syncthreads();
    for (int st = 128; st > 0; st >>= 1) { if (tid < st) red[tid] += red[tid + st]; __syncthreads(); }
    const float rstd = rsqrtf(red[0] * (1.0f / H_) + 1e-8f);
    const float* up = proj + (size_t)row * (4 * H_);   // U = first H columns
    #pragma unroll
    for (int u = 0; u < 4; ++u) {
        int c = tid + u * 256;
        float ov = (vals[u] - mean) * rstd * g[c] + bta[c];
        float uu = up[c];
        gated[(size_t)row * H_ + c] = (bf16)(uu * sigmf(uu) * ov);
    }
}

// ---------------------------------------------------------------------------
// Final layernorm -> d_out (f32)
// ---------------------------------------------------------------------------
__global__ __launch_bounds__(256)
void final_ln(const float* __restrict__ X, const float* __restrict__ g,
              const float* __restrict__ bta, float* __restrict__ outp) {
    __shared__ float red[256];
    const int row = blockIdx.x;
    const int tid = threadIdx.x;
    const float* x = X + (size_t)row * H_;
    float vals[4], s = 0.0f;
    #pragma unroll
    for (int u = 0; u < 4; ++u) { vals[u] = x[tid + u * 256]; s += vals[u]; }
    red[tid] = s; __syncthreads();
    for (int st = 128; st > 0; st >>= 1) { if (tid < st) red[tid] += red[tid + st]; __syncthreads(); }
    const float mean = red[0] * (1.0f / H_);
    __syncthreads();
    s = 0.0f;
    #pragma unroll
    for (int u = 0; u < 4; ++u) { float d = vals[u] - mean; s += d * d; }
    red[tid] = s; __syncthreads();
    for (int st = 128; st > 0; st >>= 1) { if (tid < st) red[tid] += red[tid + st]; __syncthreads(); }
    const float rstd = rsqrtf(red[0] * (1.0f / H_) + 1e-8f);
    #pragma unroll
    for (int u = 0; u < 4; ++u) {
        int c = tid + u * 256;
        outp[(size_t)row * H_ + c] = (vals[u] - mean) * rstd * g[c] + bta[c];
    }
}

__global__ void loss_mask_kernel(const int* __restrict__ nm, float* __restrict__ outp) {
    int i = blockIdx.x * blockDim.x + threadIdx.x;
    if (i < ROWS_) outp[i] = (nm[i] == 1) ? 1.0f : 0.0f;
}

// ---------------------------------------------------------------------------
// Host orchestration
// ---------------------------------------------------------------------------
extern "C" void kernel_launch(void* const* d_in, const int* in_sizes, int n_in,
                              void* d_out, int out_size, void* d_ws, size_t ws_size,
                              hipStream_t stream) {
    (void)in_sizes; (void)n_in; (void)out_size; (void)ws_size;
    const int*   user_item = (const int*)d_in[0];
    const int*   pos_seqs  = (const int*)d_in[1];
    const int*   neg_seqs  = (const int*)d_in[2];
    const int*   maskp     = (const int*)d_in[3];
    const int*   next_mask = (const int*)d_in[4];
    const float* item_emb  = (const float*)d_in[5];
    const float* user_emb  = (const float*)d_in[6];
    const float* itemdnn_w = (const float*)d_in[7];
    const float* itemdnn_b = (const float*)d_in[8];
    const float* userdnn_w = (const float*)d_in[9];
    const float* userdnn_b = (const float*)d_in[10];
    const float* proj_w    = (const float*)d_in[11];
    const float* proj_b    = (const float*)d_in[12];
    const float* ln_g      = (const float*)d_in[13];
    const float* ln_b      = (const float*)d_in[14];
    const float* tr_w      = (const float*)d_in[15];
    const float* tr_b      = (const float*)d_in[16];
    const float* last_g    = (const float*)d_in[17];
    const float* last_b    = (const float*)d_in[18];
    float* out = (float*)d_out;

    char* ws = (char*)d_ws;
    size_t off = 0;
    auto alloc = [&](size_t bytes) -> void* {
        void* p = ws + off;
        off += (bytes + 255) & ~(size_t)255;
        return p;
    };

    bf16*  w_it  = (bf16*)alloc((size_t)H_ * H_ * 2);
    bf16*  w_us  = (bf16*)alloc((size_t)H_ * H_ * 2);
    bf16*  w_pj  = (bf16*)alloc((size_t)L_ * H_ * 4 * H_ * 2);
    bf16*  w_tr  = (bf16*)alloc((size_t)L_ * H_ * H_ * 2);
    bf16*  e_a   = (bf16*)alloc((size_t)ROWS_ * H_ * 2);
    bf16*  e_b   = (bf16*)alloc((size_t)ROWS_ * H_ * 2);
    float* ai    = (float*)alloc((size_t)ROWS_ * H_ * 4);
    float* au    = (float*)alloc((size_t)ROWS_ * H_ * 4);
    float* x     = (float*)alloc((size_t)ROWS_ * H_ * 4);
    bf16*  xb    = (bf16*)alloc((size_t)ROWS_ * H_ * 2);
    float* projv = (float*)alloc((size_t)ROWS_ * 4 * H_ * 4);
    bf16*  qr    = (bf16*)alloc((size_t)ROWS_ * H_ * 2);
    bf16*  kr    = (bf16*)alloc((size_t)ROWS_ * H_ * 2);
    bf16*  vh    = (bf16*)alloc((size_t)ROWS_ * H_ * 2);
    float* ob    = (float*)alloc((size_t)ROWS_ * H_ * 4);
    bf16*  gated = (bf16*)alloc((size_t)ROWS_ * H_ * 2);

    auto cdiv = [](long n, long d) { return (int)((n + d - 1) / d); };

    // Weights -> bf16, fragment-major
    f2b_swz<<<cdiv((long)H_ * H_, 256), 256, 0, stream>>>(itemdnn_w, w_it, H_, H_);
    f2b_swz<<<cdiv((long)H_ * H_, 256), 256, 0, stream>>>(userdnn_w, w_us, H_, H_);
    for (int l = 0; l < L_; ++l) {
        f2b_swz<<<cdiv((long)H_ * 4 * H_, 256), 256, 0, stream>>>(
            proj_w + (size_t)l * H_ * 4 * H_, w_pj + (size_t)l * H_ * 4 * H_, H_, 4 * H_);
        f2b_swz<<<cdiv((long)H_ * H_, 256), 256, 0, stream>>>(
            tr_w + (size_t)l * H_ * H_, w_tr + (size_t)l * H_ * H_, H_, H_);
    }

    // feat2emb(include_user=True)
    gather_rows<<<cdiv((long)ROWS_ * H_, 256), 256, 0, stream>>>(user_item, maskp, 1, item_emb, e_a, ROWS_);
    gather_rows<<<cdiv((long)ROWS_ * H_, 256), 256, 0, stream>>>(user_item, maskp, 2, user_emb, e_b, ROWS_);
    gemm_bf16<<<dim3(H_ / 256, ROWS_ / 32), 256, 0, stream>>>(
        e_a, w_it, itemdnn_b, nullptr, ai, nullptr, ROWS_, H_, H_, 1);
    gemm_bf16<<<dim3(H_ / 256, ROWS_ / 32), 256, 0, stream>>>(
        e_b, w_us, userdnn_b, nullptr, au, nullptr, ROWS_, H_, H_, 1);
    combine_kernel<<<cdiv((long)ROWS_ * H_, 256), 256, 0, stream>>>(ai, au, x, xb);

    // HSTU layers
    for (int l = 0; l < L_; ++l) {
        gemm_bf16<<<dim3(4 * H_ / 256, ROWS_ / 32), 256, 0, stream>>>(
            xb, w_pj + (size_t)l * H_ * 4 * H_, proj_b + (size_t)l * 4 * H_,
            nullptr, projv, nullptr, ROWS_, 4 * H_, H_, 0);
        rope_split<<<cdiv((long)B_ * NH_ * S_ * 32, 256), 256, 0, stream>>>(projv, qr, kr, vh);
        hstu_attn<<<(B_ * NH_ * (S_ / 16)) / 4, 128, 0, stream>>>(qr, kr, vh, maskp, ob);
        ln_gate<<<ROWS_, 256, 0, stream>>>(ob, projv, ln_g + (size_t)l * H_, ln_b + (size_t)l * H_, gated);
        gemm_bf16<<<dim3(H_ / 256, ROWS_ / 32), 256, 0, stream>>>(
            gated, w_tr + (size_t)l * H_ * H_, tr_b + (size_t)l * H_,
            x, x, xb, ROWS_, H_, H_, 0);
    }
    final_ln<<<ROWS_, 256, 0, stream>>>(x, last_g, last_b, out);

    // pos / neg embeddings (reuse e_a / e_b after main path has consumed them)
    gather_rows<<<cdiv((long)ROWS_ * H_, 256), 256, 0, stream>>>(pos_seqs, nullptr, 0, item_emb, e_a, ROWS_);
    gather_rows<<<cdiv((long)ROWS_ * H_, 256), 256, 0, stream>>>(neg_seqs, nullptr, 0, item_emb, e_b, ROWS_);
    gemm_bf16<<<dim3(H_ / 256, ROWS_ / 32), 256, 0, stream>>>(
        e_a, w_it, itemdnn_b, nullptr, out + (size_t)ROWS_ * H_, nullptr, ROWS_, H_, H_, 1);
    gemm_bf16<<<dim3(H_ / 256, ROWS_ / 32), 256, 0, stream>>>(
        e_b, w_it, itemdnn_b, nullptr, out + 2 * (size_t)ROWS_ * H_, nullptr, ROWS_, H_, H_, 1);

    loss_mask_kernel<<<cdiv(ROWS_, 256), 256, 0, stream>>>(next_mask, out + 3 * (size_t)ROWS_ * H_);
}
